// QuantizedLinear_71880572666115
// MI455X (gfx1250) — matrix operand order repaired and checked by reading, not compiled, not run on hardware
//
#include <hip/hip_runtime.h>

#define M_TOT 8192   // B*S = 4*2048
#define K_TOT 4096   // D_IN
#define N_TOT 4096   // D_OUT

typedef __attribute__((ext_vector_type(16))) __bf16 v16bf;
typedef __attribute__((ext_vector_type(8)))  __bf16 v8bf;
typedef __attribute__((ext_vector_type(8)))  float  v8f;

// ---------------- fp32 -> bf16 (RNE) helpers ----------------
__device__ __forceinline__ unsigned int f2bf(float f) {
    unsigned int u = __float_as_uint(f);
    return (u + 0x7FFFu + ((u >> 16) & 1u)) >> 16;
}
__device__ __forceinline__ unsigned int pack2(float lo, float hi) {
    return f2bf(lo) | (f2bf(hi) << 16);
}

// x [M,K] fp32 -> bf16 ; each thread converts 8 elements
__global__ void cvt_x_bf16(const float* __restrict__ x, uint4* __restrict__ out) {
    size_t i = (size_t)blockIdx.x * blockDim.x + threadIdx.x;
    const float4* xv = (const float4*)x;
    float4 a = xv[2 * i], b = xv[2 * i + 1];
    uint4 r;
    r.x = pack2(a.x, a.y); r.y = pack2(a.z, a.w);
    r.z = pack2(b.x, b.y); r.w = pack2(b.z, b.w);
    out[i] = r;
}

// w [N,K] int32 codes in [-128,127] -> bf16 (exact)
__global__ void cvt_w_bf16(const int* __restrict__ w, uint4* __restrict__ out) {
    size_t i = (size_t)blockIdx.x * blockDim.x + threadIdx.x;
    const int4* wv = (const int4*)w;
    int4 a = wv[2 * i], b = wv[2 * i + 1];
    uint4 r;
    r.x = pack2((float)a.x, (float)a.y); r.y = pack2((float)a.z, (float)a.w);
    r.z = pack2((float)b.x, (float)b.y); r.w = pack2((float)b.z, (float)b.w);
    out[i] = r;
}

// rowsum[m] = sum_k x[m,k]  (fp32 accuracy), one block per row
__global__ void rowsum_kernel(const float* __restrict__ x, float* __restrict__ rs) {
    __shared__ float red[256];
    const int row = blockIdx.x;
    const float4* xr = (const float4*)(x + (size_t)row * K_TOT);
    float s = 0.f;
    for (int k = threadIdx.x; k < K_TOT / 4; k += 256) {
        float4 v = xr[k];
        s += v.x + v.y + v.z + v.w;
    }
    red[threadIdx.x] = s;
    __syncthreads();
    for (int off = 128; off > 0; off >>= 1) {
        if ((int)threadIdx.x < off) red[threadIdx.x] += red[threadIdx.x + off];
        __syncthreads();
    }
    if (threadIdx.x == 0) rs[row] = red[0];
}

// ---------------- CDNA5 async global->LDS copy (ASYNCcnt path) ----------------
// For generic pointers into the LDS aperture, flat_addr[31:0] is the LDS byte
// offset (ISA §10.2), so truncating the shared-memory pointer yields VDST.
__device__ __forceinline__ void async_copy_b128(const unsigned short* g, unsigned short* l) {
    unsigned int loff = (unsigned int)(unsigned long long)l;
    asm volatile("global_load_async_to_lds_b128 %0, %1, off"
                 :: "v"(loff), "v"(g)
                 : "memory");
}
__device__ __forceinline__ void wait_asynccnt0() {
    asm volatile("s_wait_asynccnt 0x0" ::: "memory");
}

// ---------------- WMMA GEMM ----------------
#define BM 128
#define BN 128
#define BK 64
#define LDST 72                 // LDS row stride in bf16 elements (144B, 16B aligned)
#define SDS (2 * BM * LDST)     // one stage: A tile (BM*LDST) + B tile (BM*LDST)

__launch_bounds__(256, 2)
__global__ void qlinear_wmma(const unsigned short* __restrict__ A,   // x_bf16 [M,K]
                             const unsigned short* __restrict__ Bw,  // w_bf16 [N,K]
                             const float* __restrict__ rowsum,
                             const float* __restrict__ scale,
                             const float* __restrict__ zp,
                             const float* __restrict__ bias,
                             float* __restrict__ out) {
    __shared__ __align__(16) unsigned short smem[2 * SDS];   // 73,728 B double-buffered

    const int tid  = threadIdx.x;
    const int lane = tid & 31;
    const int wid  = tid >> 5;
    const int lo   = lane & 15;     // M row (A) / N col (B) within 16-tile
    const int hf   = lane >> 4;     // lane-half selects K sub-chunk
    const int waveM = (wid >> 2) * 64;   // waves 2 (M) x 4 (N)
    const int waveN = (wid & 3) * 32;
    const int blockN = blockIdx.x * BN;
    const int blockM = blockIdx.y * BM;

    v8f acc[4][2];
    #pragma unroll
    for (int mt = 0; mt < 4; ++mt)
        #pragma unroll
        for (int nt = 0; nt < 2; ++nt)
            acc[mt][nt] = (v8f)(0.f);

    // global->LDS staging: 256 threads, each issues 4 x b128 async copies per tile
    const int r0 = tid >> 3;          // 0..31 (row group)
    const int c8 = (tid & 7) * 8;     // 0..56 (col, bf16 elements)
    const unsigned short* gA = A  + (size_t)blockM * K_TOT;
    const unsigned short* gB = Bw + (size_t)blockN * K_TOT;

    // prologue: fill stage 0
    #pragma unroll
    for (int i = 0; i < 4; ++i) {
        int row = r0 + i * 32;
        async_copy_b128(gA + (size_t)row * K_TOT + c8, &smem[row * LDST + c8]);
        async_copy_b128(gB + (size_t)row * K_TOT + c8, &smem[BM * LDST + row * LDST + c8]);
    }
    wait_asynccnt0();
    __syncthreads();

    int s = 0;
    const int NKB = K_TOT / BK;   // 64 stages
    for (int kb = 0; kb < NKB; ++kb) {
        if (kb + 1 < NKB) {
            const int koff = (kb + 1) * BK;
            const int ns = s ^ 1;
            #pragma unroll
            for (int i = 0; i < 4; ++i) {
                int row = r0 + i * 32;
                async_copy_b128(gA + (size_t)row * K_TOT + koff + c8,
                                &smem[ns * SDS + row * LDST + c8]);
                async_copy_b128(gB + (size_t)row * K_TOT + koff + c8,
                                &smem[ns * SDS + BM * LDST + row * LDST + c8]);
            }
        }
        const unsigned short* sA = &smem[s * SDS];
        const unsigned short* sB = &smem[s * SDS + BM * LDST];
        #pragma unroll
        for (int ko = 0; ko < BK; ko += 32) {
            v16bf af[4], bfr[2];
            // A fragment (16x32): lane lo = row; low lanes K {0..7,16..23}, high {8..15,24..31}
            #pragma unroll
            for (int mt = 0; mt < 4; ++mt) {
                int row = waveM + mt * 16 + lo;
                const __bf16* p = (const __bf16*)&sA[row * LDST + ko + hf * 8];
                v8bf c0 = *(const v8bf*)p;
                v8bf c1 = *(const v8bf*)(p + 16);
                af[mt] = __builtin_shufflevector(c0, c1, 0,1,2,3,4,5,6,7,8,9,10,11,12,13,14,15);
            }
            // B fragment (32x16): lane lo = col; low lanes K {0..7,16..23}, high {8..15,24..31}
            #pragma unroll
            for (int nt = 0; nt < 2; ++nt) {
                int col = waveN + nt * 16 + lo;
                const __bf16* p = (const __bf16*)&sB[col * LDST + ko + hf * 8];
                v8bf c0 = *(const v8bf*)p;
                v8bf c1 = *(const v8bf*)(p + 16);
                bfr[nt] = __builtin_shufflevector(c0, c1, 0,1,2,3,4,5,6,7,8,9,10,11,12,13,14,15);
            }
            #pragma unroll
            for (int mt = 0; mt < 4; ++mt)
                #pragma unroll
                for (int nt = 0; nt < 2; ++nt)
                    acc[mt][nt] = __builtin_amdgcn_wmma_f32_16x16x32_bf16(
                        false, af[mt], false, bfr[nt], (short)0, acc[mt][nt], false, false);
        }
        wait_asynccnt0();
        __syncthreads();
        s ^= 1;
    }

    // epilogue: out = scale*acc + bias - scale*zp*rowsum
    // C/D layout: VGPR r -> M = r (lanes 0-15) / r+8 (lanes 16-31); N = lane&15
    #pragma unroll
    for (int mt = 0; mt < 4; ++mt) {
        float rs[8];
        #pragma unroll
        for (int r = 0; r < 8; ++r)
            rs[r] = rowsum[blockM + waveM + mt * 16 + hf * 8 + r];
        #pragma unroll
        for (int nt = 0; nt < 2; ++nt) {
            const int o = blockN + waveN + nt * 16 + lo;
            const float sc = scale[o];
            const float sz = sc * zp[o];
            const float bi = bias[o];
            #pragma unroll
            for (int r = 0; r < 8; ++r) {
                const int m = blockM + waveM + mt * 16 + hf * 8 + r;
                out[(size_t)m * N_TOT + o] = sc * acc[mt][nt][r] + (bi - sz * rs[r]);
            }
        }
    }
}

extern "C" void kernel_launch(void* const* d_in, const int* in_sizes, int n_in,
                              void* d_out, int out_size, void* d_ws, size_t ws_size,
                              hipStream_t stream) {
    const float* x     = (const float*)d_in[0];
    const int*   w     = (const int*)d_in[1];
    const float* scale = (const float*)d_in[2];
    const float* zp    = (const float*)d_in[3];
    const float* bias  = (const float*)d_in[4];
    float* out = (float*)d_out;

    unsigned short* xbf = (unsigned short*)d_ws;                     // 64 MB
    unsigned short* wbf = xbf + (size_t)M_TOT * K_TOT;               // +32 MB
    float* rowsum = (float*)(wbf + (size_t)N_TOT * K_TOT);           // +32 KB

    cvt_x_bf16<<<(int)(((size_t)M_TOT * K_TOT) / (256 * 8)), 256, 0, stream>>>(x, (uint4*)xbf);
    cvt_w_bf16<<<(int)(((size_t)N_TOT * K_TOT) / (256 * 8)), 256, 0, stream>>>(w, (uint4*)wbf);
    rowsum_kernel<<<M_TOT, 256, 0, stream>>>(x, rowsum);

    dim3 grid(N_TOT / BN, M_TOT / BM);   // 32 x 64 workgroups
    qlinear_wmma<<<grid, 256, 0, stream>>>(xbf, wbf, rowsum, scale, zp, bias, out);
}
